// MultiHeadAttention_66434554134753
// MI455X (gfx1250) — matrix operand order repaired and checked by reading, not compiled, not run on hardware
//
#include <hip/hip_runtime.h>
#include <hip/hip_bf16.h>
#include <math.h>
#include <stdint.h>

// MI455X (gfx1250) MHA forward.
// d_out = [out (2*2048*1024 f32) | attn_weights (2*16*2048*2048 f32)]
// d_ws  = Qp [B,H,S,DK] | Kp [B,H,S,DK] | Vt [B,H,DK,S] | ctx [B,S,D], bf16.

#define D_MODEL 1024
#define N_HEADS 16
#define DK      64
#define BATCH   2
#define SEQ     2048
#define NEGINF  -1000000000.0f

typedef __attribute__((ext_vector_type(16))) __bf16 bf16x16;
typedef __attribute__((ext_vector_type(8)))  float  f32x8;
typedef __attribute__((ext_vector_type(4)))  unsigned uint4v;
typedef __attribute__((ext_vector_type(8)))  unsigned uint8v;

union FragB16 { bf16x16 v; unsigned int w[8]; unsigned short u[16]; };

__device__ __forceinline__ unsigned short f2bf(float f) {
  union { float f; unsigned u; } x; x.f = f;
  unsigned r = x.u + 0x7FFFu + ((x.u >> 16) & 1u);   // round-to-nearest-even
  return (unsigned short)(r >> 16);
}

#if __has_builtin(__builtin_amdgcn_cvt_pk_bf16_f32)
typedef __attribute__((ext_vector_type(2))) __bf16 bf16x2;
__device__ __forceinline__ unsigned pk2bf(float lo, float hi) {
  union { bf16x2 v; unsigned u; } x;
  x.v = __builtin_amdgcn_cvt_pk_bf16_f32(lo, hi);
  return x.u;
}
#else
__device__ __forceinline__ unsigned pk2bf(float lo, float hi) {
  return (unsigned)f2bf(lo) | ((unsigned)f2bf(hi) << 16);
}
#endif

// A-matrix (16x32 bf16) K-pair base for VGPR v, lane-half h (ISA 7.12.2)
__device__ __forceinline__ int a_kbase(int v, int h) {
  return (v < 4 ? 2 * v : 16 + 2 * (v - 4)) + 8 * h;
}
// B-matrix (32x16 bf16) K-pair base for VGPR v, lane-half h (same K order as A)
__device__ __forceinline__ int b_kbase(int v, int h) {
  return (v < 4 ? 2 * v : 16 + 2 * (v - 4)) + 8 * h;
}

__device__ __forceinline__ f32x8 wmma_bf16(bf16x16 a, bf16x16 b, f32x8 c) {
  return __builtin_amdgcn_wmma_f32_16x16x32_bf16(false, a, false, b, (short)0, c,
                                                 false, false);
}

// ---------------------------------------------------------------------------
// Projection GEMM: Y[M=4096, N=1024] = X[4096,1024] @ W^T[1024,1024] + bias
// Block = 256 threads (8 waves), tile 128x128, LDS-staged bf16 operands.
// X_IS_F32: X is fp32 (converted on stage) else bf16 (ctx).
// OMODE: 0 = fp32 row-major [4096,1024]
//        1 = bf16 head-major   [B,H,S,DK]   (Q, K)
//        2 = bf16 head-transp. [B,H,DK,S]   (V)
// ---------------------------------------------------------------------------
template <bool X_IS_F32, int OMODE>
__global__ __launch_bounds__(256) void proj_gemm(const void* __restrict__ Xv,
                                                 const float* __restrict__ W,
                                                 const float* __restrict__ bias,
                                                 void* __restrict__ Yv) {
  __shared__ unsigned short Xs[128][32];
  __shared__ unsigned short Ws[128][32];

  const int tid  = threadIdx.x;
  const int lane = tid & 31;
  const int wv   = tid >> 5;                 // 0..7 waves
  const int bn   = blockIdx.x & 7;           // N/128 = 8
  const int bm   = blockIdx.x >> 3;          // M/128 = 32
  const int m0   = bm * 128, n0 = bn * 128;
  const int mw   = m0 + wv * 16;

  f32x8 acc[8] = {};

  const int srow  = tid >> 1;                // 0..127
  const int skoff = (tid & 1) * 16;          // 0 / 16

  for (int kk = 0; kk < 1024; kk += 32) {
    __syncthreads();
    if (X_IS_F32) {
      const float* src = (const float*)Xv + (size_t)(m0 + srow) * 1024 + kk + skoff;
      __builtin_prefetch(src + 128, 0, 0);   // 4 k-steps ahead
#pragma unroll
      for (int j = 0; j < 16; j += 2)
        *(unsigned*)&Xs[srow][skoff + j] = pk2bf(src[j], src[j + 1]);
    } else {
      const unsigned short* src =
          (const unsigned short*)Xv + (size_t)(m0 + srow) * 1024 + kk + skoff;
      __builtin_prefetch(src + 128, 0, 0);
#pragma unroll
      for (int j = 0; j < 16; ++j) Xs[srow][skoff + j] = src[j];
    }
    {
      const float* src = W + (size_t)(n0 + srow) * 1024 + kk + skoff;
      __builtin_prefetch(src + 128, 0, 0);
#pragma unroll
      for (int j = 0; j < 16; j += 2)
        *(unsigned*)&Ws[srow][skoff + j] = pk2bf(src[j], src[j + 1]);
    }
    __syncthreads();

    FragB16 a;
    const int am = wv * 16 + (lane & 15);
    const int ah = lane >> 4;
#pragma unroll
    for (int v = 0; v < 8; ++v)
      a.w[v] = *(const unsigned int*)&Xs[am][a_kbase(v, ah)];

    const int bnl = lane & 15, bh = lane >> 4;
#pragma unroll
    for (int t = 0; t < 8; ++t) {
      FragB16 bf;
#pragma unroll
      for (int v = 0; v < 8; ++v)
        bf.w[v] = *(const unsigned int*)&Ws[t * 16 + bnl][b_kbase(v, bh)];
      acc[t] = wmma_bf16(a.v, bf.v, acc[t]);
    }
  }

  const int colh = lane >> 4;
#pragma unroll
  for (int t = 0; t < 8; ++t) {
    const int j  = n0 + t * 16 + (lane & 15);
    const float bj = bias[j];
#pragma unroll
    for (int r = 0; r < 8; ++r) {
      const int i = mw + r + 8 * colh;
      const float y = acc[t][r] + bj;
      if (OMODE == 1) {
        const int b = i >> 11, s = i & (SEQ - 1);
        const int hh = j >> 6, dd = j & 63;
        ((unsigned short*)Yv)[(((size_t)(b * N_HEADS + hh) * SEQ) + s) * DK + dd] =
            f2bf(y);
      } else if (OMODE == 2) {
        const int b = i >> 11, s = i & (SEQ - 1);
        const int hh = j >> 6, dd = j & 63;
        ((unsigned short*)Yv)[(((size_t)(b * N_HEADS + hh) * DK) + dd) * SEQ + s] =
            f2bf(y);
      } else {
        ((float*)Yv)[(size_t)i * 1024 + j] = y;
      }
    }
  }
}

// ---------------------------------------------------------------------------
// Attention: one workgroup (4 waves) per (b, h, 16-query-row tile).
// Q tile DMA'd to LDS via Tensor Data Mover; scores -> 16x2048 f32 LDS tile ->
// softmax -> f32 weights to d_out (coalesced float4) -> P@V (V transposed,
// contiguous fragments) -> bf16 ctx.
// ---------------------------------------------------------------------------
__global__ __launch_bounds__(128) void attn_kernel(
    const unsigned short* __restrict__ Qp, const unsigned short* __restrict__ Kp,
    const unsigned short* __restrict__ Vp, float* __restrict__ attnW,
    unsigned short* __restrict__ ctx) {
  __shared__ float sc[16 * SEQ];             // 128 KB (WGP LDS = 320 KB)
  __shared__ unsigned short sh_q[16 * DK];   // 2 KB Q tile (TDM destination)
  __shared__ float red[16][8];
  __shared__ float rmax[16], rsum[16];

  const int tid  = threadIdx.x;
  const int lane = tid & 31;
  const int wv   = tid >> 5;                 // 0..3
  const int bid  = blockIdx.x;
  const int qt   = bid & 127;                // query tile
  const int hh   = (bid >> 7) & 15;
  const int b    = bid >> 11;
  const int q0   = qt * 16;
  const int nvalid = 16 * (qt + 1);          // causal: cols beyond are all zero

  const unsigned short* Qh = Qp + (size_t)(b * N_HEADS + hh) * SEQ * DK;
  const unsigned short* Kh = Kp + (size_t)(b * N_HEADS + hh) * SEQ * DK;
  const unsigned short* Vt = Vp + (size_t)(b * N_HEADS + hh) * DK * SEQ;

  // --- TDM: DMA the 16x64 bf16 Q tile into LDS (wave 0 issues; EXEC ignored,
  // scalar branch keeps other waves away). 2-D tile -> 2 SGPR groups.
  if (wv == 0) {
    const unsigned long long ga = (unsigned long long)(uintptr_t)(Qh + (size_t)q0 * DK);
    const unsigned lds_off = (unsigned)(uintptr_t)(void*)sh_q;
    uint4v g0;
    g0.x = 1u;                                         // count=1, user descriptor
    g0.y = lds_off;                                    // lds_addr
    g0.z = (unsigned)(ga & 0xFFFFFFFFu);               // global_addr[31:0]
    g0.w = (unsigned)((ga >> 32) & 0x1FFFFFFu) | (2u << 30);  // addr[56:32], type=2
    uint8v g1;
    g1[0] = (1u << 16);        // data_size=1 (2 bytes/elem)
    g1[1] = (64u << 16);       // tensor_dim0 = 64 (elems per row)
    g1[2] = ((unsigned)SEQ << 16);  // tensor_dim1 = 2048 rows
    g1[3] = (64u << 16);       // tile_dim0 = 64
    g1[4] = 16u;               // tile_dim1 = 16 rows, tile_dim2 = 0
    g1[5] = 64u;               // tensor_dim0_stride = 64 (lo 32)
    g1[6] = 0u;                // stride hi | tensor_dim1_stride lo
    g1[7] = 0u;
    asm volatile("tensor_load_to_lds %0, %1" :: "s"(g0), "s"(g1) : "memory");
    __builtin_amdgcn_s_wait_tensorcnt(0);
  }
  __syncthreads();

  const int am = lane & 15, ah = lane >> 4;

  // Q fragments for k=0..31 and k=32..63 (from LDS, reused for all col tiles)
  FragB16 a0, a1;
#pragma unroll
  for (int v = 0; v < 8; ++v) {
    const int kb = a_kbase(v, ah);
    a0.w[v] = *(const unsigned int*)&sh_q[am * DK + kb];
    a1.w[v] = *(const unsigned int*)&sh_q[am * DK + 32 + kb];
  }

  // Phase 1: S = (Q K^T) / sqrt(dk), causal mask, into LDS
  for (int ct = wv; ct <= qt; ct += 4) {
    FragB16 b0, b1;
    const int key = ct * 16 + (lane & 15);
    const unsigned short* kr = Kh + (size_t)key * DK;
#pragma unroll
    for (int v = 0; v < 8; ++v) {
      const int kb = b_kbase(v, lane >> 4);
      b0.w[v] = *(const unsigned int*)(kr + kb);
      b1.w[v] = *(const unsigned int*)(kr + 32 + kb);
    }
    f32x8 acc = {};
    acc = wmma_bf16(a0.v, b0.v, acc);
    acc = wmma_bf16(a1.v, b1.v, acc);

    const int cg = ct * 16 + (lane & 15);
#pragma unroll
    for (int r = 0; r < 8; ++r) {
      const int rl = r + 8 * (lane >> 4);
      float s = acc[r] * 0.125f;             // 1/sqrt(64)
      if (cg > q0 + rl) s = NEGINF;          // causal mask
      sc[rl * SEQ + cg] = s;
    }
  }
  __syncthreads();

  // Phase 2: per-row max and sum(exp)
  {
    const int row = tid & 15, seg = tid >> 4;
    const int chunk = nvalid >> 3;
    const int c0 = seg * chunk, c1 = c0 + chunk;
    float mx = -3.4e38f;
    for (int c = c0; c < c1; ++c) mx = fmaxf(mx, sc[row * SEQ + c]);
    red[row][seg] = mx;
    __syncthreads();
    if (tid < 16) {
      float m = red[tid][0];
#pragma unroll
      for (int s = 1; s < 8; ++s) m = fmaxf(m, red[tid][s]);
      rmax[tid] = m;
    }
    __syncthreads();
    const float rm = rmax[row];
    float sum = 0.0f;
    for (int c = c0; c < c1; ++c) sum += __expf(sc[row * SEQ + c] - rm);
    red[row][seg] = sum;
    __syncthreads();
    if (tid < 16) {
      float s = 0.0f;
#pragma unroll
      for (int q = 0; q < 8; ++q) s += red[tid][q];
      rsum[tid] = s;
    }
    __syncthreads();
  }

  // Phase 3: normalize, write weights (float4, coalesced), keep P in LDS
  {
    const int row = tid >> 3, j8 = tid & 7;
    const float mx = rmax[row], inv = 1.0f / rsum[row];
    float* dst = attnW + ((size_t)((b * N_HEADS + hh) * SEQ) + q0 + row) * SEQ;
    for (int c = j8 * 4; c < SEQ; c += 32) {
      float4 o = make_float4(0.f, 0.f, 0.f, 0.f);
      if (c < nvalid) {                      // nvalid multiple of 16
        o.x = __expf(sc[row * SEQ + c]     - mx) * inv;
        o.y = __expf(sc[row * SEQ + c + 1] - mx) * inv;
        o.z = __expf(sc[row * SEQ + c + 2] - mx) * inv;
        o.w = __expf(sc[row * SEQ + c + 3] - mx) * inv;
        sc[row * SEQ + c]     = o.x;
        sc[row * SEQ + c + 1] = o.y;
        sc[row * SEQ + c + 2] = o.z;
        sc[row * SEQ + c + 3] = o.w;
      }
      *(float4*)(dst + c) = o;
    }
  }
  __syncthreads();

  // Phase 4: O = P @ V ; wave wv owns head-dim cols [16*wv, 16*wv+15].
  // V is transposed [DK, SEQ]: K-pairs contiguous -> b128-mergeable loads.
  {
    f32x8 acc = {};
    const int d  = wv * 16 + (lane & 15);
    const int h2 = lane >> 4;
    const unsigned short* vrow = Vt + (size_t)d * SEQ;
    const float* prow = &sc[(lane & 15) * SEQ];
    const int nchunks = (nvalid + 31) / 32;
    for (int kc = 0; kc < nchunks; ++kc) {
      FragB16 pa, vb;
      const int kA = kc * 32;
      if (kA + 32 <= nvalid) {               // full chunk: unguarded, vectorizable
#pragma unroll
        for (int v = 0; v < 8; ++v) {
          const int kb = kA + a_kbase(v, h2);
          pa.w[v] = pk2bf(prow[kb], prow[kb + 1]);
          vb.w[v] = *(const unsigned int*)(vrow + kA + b_kbase(v, h2));
        }
      } else {                               // partial tail chunk
#pragma unroll
        for (int v = 0; v < 8; ++v) {
          const int kb = kA + a_kbase(v, h2);
          pa.w[v] = (kb < nvalid) ? pk2bf(prow[kb], prow[kb + 1]) : 0u;
          const int kb2 = kA + b_kbase(v, h2);
          vb.w[v] = (kb2 < nvalid) ? *(const unsigned int*)(vrow + kb2) : 0u;
        }
      }
      acc = wmma_bf16(pa.v, vb.v, acc);
    }
#pragma unroll
    for (int r = 0; r < 8; ++r) {
      const int s = q0 + r + 8 * (lane >> 4);
      ctx[(size_t)(b * SEQ + s) * D_MODEL + hh * DK + d] = f2bf(acc[r]);
    }
  }
}

// ---------------------------------------------------------------------------
extern "C" void kernel_launch(void* const* d_in, const int* in_sizes, int n_in,
                              void* d_out, int out_size, void* d_ws, size_t ws_size,
                              hipStream_t stream) {
  (void)in_sizes; (void)n_in; (void)out_size; (void)ws_size;
  const float* q  = (const float*)d_in[0];
  const float* k  = (const float*)d_in[1];
  const float* v  = (const float*)d_in[2];
  // d_in[3] = causal mask (implicit in the attention kernel)
  const float* wq = (const float*)d_in[4];
  const float* bq = (const float*)d_in[5];
  const float* wk = (const float*)d_in[6];
  const float* bk = (const float*)d_in[7];
  const float* wvw = (const float*)d_in[8];
  const float* bv = (const float*)d_in[9];
  const float* wo = (const float*)d_in[10];
  const float* bo = (const float*)d_in[11];

  float* out   = (float*)d_out;
  float* attnW = out + (size_t)BATCH * SEQ * D_MODEL;

  const size_t NE = (size_t)BATCH * SEQ * D_MODEL;     // 4,194,304 elems
  unsigned short* Qp  = (unsigned short*)d_ws;
  unsigned short* Kp  = Qp + NE;
  unsigned short* Vt  = Kp + NE;                        // transposed [B,H,DK,S]
  unsigned short* ctx = Vt + NE;                        // total 32 MB in ws

  const dim3 gGrid(256), gBlk(256);
  proj_gemm<true,  1><<<gGrid, gBlk, 0, stream>>>(q,   wq,  bq, Qp);
  proj_gemm<true,  1><<<gGrid, gBlk, 0, stream>>>(k,   wk,  bk, Kp);
  proj_gemm<true,  2><<<gGrid, gBlk, 0, stream>>>(v,   wvw, bv, Vt);
  attn_kernel<<<dim3(BATCH * N_HEADS * (SEQ / 16)), dim3(128), 0, stream>>>(
      Qp, Kp, Vt, attnW, ctx);
  proj_gemm<false, 0><<<gGrid, gBlk, 0, stream>>>(ctx, wo, bo, out);
}
